// Models_12524124635406
// MI455X (gfx1250) — matrix operand order repaired and checked by reading, not compiled, not run on hardware
//
#include <hip/hip_runtime.h>

#define NNODE   256
#define DIM     64
#define FDIM    63
#define THREADS 256
#define WAVES   8

typedef __bf16 bf16_t;
typedef __attribute__((ext_vector_type(16))) __bf16 v16bf;
typedef __attribute__((ext_vector_type(8)))  float  v8f;
typedef __attribute__((ext_vector_type(4)))  unsigned int v4u;
typedef __attribute__((ext_vector_type(8)))  int v8i;
typedef __attribute__((ext_vector_type(4)))  int v4i;

// Dynamic LDS layout (bytes):
//   x0    : [      0,  32768)  256x64 bf16 (padded layer-1 input)
//   h     : [  32768,  65536)  256x64 bf16 (layer-1 output / layer-2 input)
//   stage : [  65536, 131072)  2 x (32 rows x 256 col f32) adj TDM staging
//   agg   : [ 131072, 139264)  2 x (32x64 bf16) agg scratch (iteration parity)
//   y0    : [ 139264, 139520)  64 f32 (layer-2 output row 0)
#define X0_OFF   0
#define H_OFF    32768
#define STG_OFF  65536
#define AGG_OFF  131072
#define Y0_OFF   139264
#define SMEM_BYTES 139520

__device__ __forceinline__ v8f wmma_bf16(v16bf a, v16bf b, v8f c) {
    return __builtin_amdgcn_wmma_f32_16x16x32_bf16(
        /*neg_a=*/false, a, /*neg_b=*/false, b,
        /*c_mod=*/(short)0, c, /*reuse_a=*/false, /*reuse_b=*/false);
}

// ---- Tensor Data Mover: 2D f32 tile (tile_rows x 256) global -> LDS ----
// D# built per CDNA5 ISA §8.3/8.4: count=1, type=2("image"), data_size=4B,
// tensor 256x256 row-major (stride0 = 256 elems), tile_dim0=256, tile_dim1=rows.
// This toolchain exposes the 6-arg builtin: (v4u g0, v8i g1, v4i g2, v4i g3, v8i gx, i32 cpol).
__device__ __forceinline__ void tdm_load_adj_tile(const float* gsrc, void* ldst,
                                                  unsigned int tile_rows) {
    unsigned long long ga = (unsigned long long)(uintptr_t)gsrc;
    unsigned int lds_off = (unsigned int)(uintptr_t)ldst;  // addr[31:0] = LDS byte offset
    v4u g0;
    g0[0] = 1u;                                            // count=1, user mode, no gather
    g0[1] = lds_off;                                       // lds_addr
    g0[2] = (unsigned int)(ga & 0xFFFFFFFFull);            // global_addr[31:0]
    g0[3] = (unsigned int)((ga >> 32) & 0x01FFFFFFull)     // global_addr[56:32]
          | (2u << 30);                                    // type = 2
    v8i g1;
    g1[0] = (int)(2u << 16);                               // data_size = 4 bytes
    g1[1] = (int)((NNODE & 0xFFFFu) << 16);                // tensor_dim0[15:0] = 256
    g1[2] = (int)((NNODE & 0xFFFFu) << 16);                // dim0[31:16]=0 | tensor_dim1[15:0]=256
    g1[3] = (int)((NNODE & 0xFFFFu) << 16);                // dim1[31:16]=0 | tile_dim0 = 256
    g1[4] = (int)(tile_rows & 0xFFFFu);                    // tile_dim1 = rows, tile_dim2 = 0 (2D)
    g1[5] = NNODE;                                         // tensor_dim0_stride[31:0] = 256
    g1[6] = 0;                                             // stride0[47:32] | stride1[15:0]
    g1[7] = 0;                                             // stride1[47:16]
    v4i gz4 = {0, 0, 0, 0};
    v8i gz8 = {0, 0, 0, 0, 0, 0, 0, 0};
    __builtin_amdgcn_tensor_load_to_lds(g0, g1, gz4, gz4, gz8, 0);
}

// A-fragment (16x32) from bf16 LDS, row-major, stride DIM halfs.
__device__ __forceinline__ v16bf load_a_lds(const bf16_t* base, int row0, int k0, int lane) {
    int l = lane & 15, hi = lane >> 4;
    const bf16_t* p = base + (row0 + l) * DIM + k0 + hi * 8;
    v16bf a;
    ((uint4*)&a)[0] = *(const uint4*)(p);        // K +0..7  (or +8..15)
    ((uint4*)&a)[1] = *(const uint4*)(p + 16);   // K +16..23 (or +24..31)
    return a;
}

// B-fragment (32x16) from bf16 LDS, row-major K x 64.
__device__ __forceinline__ v16bf load_b_lds(const bf16_t* base, int k0, int n0, int lane) {
    int l = lane & 15, hi = lane >> 4;
    const bf16_t* p = base + (k0 + hi * 8) * DIM + n0 + l;
    v16bf b;
    #pragma unroll
    for (int s = 0; s < 8; ++s) {
        b[s]     = p[s * DIM];
        b[s + 8] = p[(s + 16) * DIM];
    }
    return b;
}

__device__ __forceinline__ v16bf pack16(float4 f0, float4 f1, float4 f2, float4 f3) {
    v16bf a;
    a[0]=(bf16_t)f0.x;  a[1]=(bf16_t)f0.y;  a[2]=(bf16_t)f0.z;  a[3]=(bf16_t)f0.w;
    a[4]=(bf16_t)f1.x;  a[5]=(bf16_t)f1.y;  a[6]=(bf16_t)f1.z;  a[7]=(bf16_t)f1.w;
    a[8]=(bf16_t)f2.x;  a[9]=(bf16_t)f2.y;  a[10]=(bf16_t)f2.z; a[11]=(bf16_t)f2.w;
    a[12]=(bf16_t)f3.x; a[13]=(bf16_t)f3.y; a[14]=(bf16_t)f3.z; a[15]=(bf16_t)f3.w;
    return a;
}

// A-fragment (16x32) from f32 LDS staging (row stride 256 floats), cvt -> bf16.
__device__ __forceinline__ v16bf load_a_stage_f32(const float* base, int row0, int k0, int lane) {
    int l = lane & 15, hi = lane >> 4;
    const float* p = base + (row0 + l) * NNODE + k0 + hi * 8;
    return pack16(*(const float4*)(p),      *(const float4*)(p + 4),
                  *(const float4*)(p + 16), *(const float4*)(p + 20));
}

// A-fragment (16x32) directly from global f32 (layer-2 adj rows 0..15, L2-hot).
__device__ __forceinline__ v16bf load_a_global_f32(const float* base, int row0, int k0,
                                                   int stride, int lane) {
    int l = lane & 15, hi = lane >> 4;
    const float* p = base + (size_t)(row0 + l) * stride + k0 + hi * 8;
    return pack16(*(const float4*)(p),      *(const float4*)(p + 4),
                  *(const float4*)(p + 16), *(const float4*)(p + 20));
}

// B-fragment (32x16) from global f32 weight [K x 64]; kmax clamps the 63-row weights
// (the padded A column is exactly zero, so the clamped row contributes nothing).
__device__ __forceinline__ v16bf load_b_global_f32(const float* w, int k0, int n0,
                                                   int kmax, int lane) {
    int l = lane & 15, hi = lane >> 4;
    v16bf b;
    #pragma unroll
    for (int s = 0; s < 16; ++s) {
        int k = k0 + hi * 8 + (s < 8 ? s : s + 8);
        if (k > kmax) k = kmax;
        b[s] = (bf16_t)w[k * DIM + n0 + l];
    }
    return b;
}

// C/D (16x16 f32): VGPR v = row M=v (lanes 0-15) or M=v+8 (lanes 16-31), N=lane&15.
__device__ __forceinline__ void store_c_bf16(bf16_t* base, int row0, int n0, v8f c, int lane) {
    int l = lane & 15, mofs = (lane >> 4) * 8;
    #pragma unroll
    for (int v = 0; v < 8; ++v)
        base[(row0 + mofs + v) * DIM + n0 + l] = (bf16_t)c[v];
}

__global__ void __launch_bounds__(THREADS)
gnn2_tdm_wmma_kernel(const float* __restrict__ adj,  const float* __restrict__ vec,
                     const float* __restrict__ w1r,  const float* __restrict__ w1t,
                     const float* __restrict__ b1,   const float* __restrict__ w2r,
                     const float* __restrict__ w2t,  const float* __restrict__ b2,
                     const float* __restrict__ wf,   const float* __restrict__ bfv,
                     float* __restrict__ out) {
    extern __shared__ float4 smem4[];
    char* smem = (char*)smem4;
    bf16_t* x0   = (bf16_t*)(smem + X0_OFF);
    bf16_t* h    = (bf16_t*)(smem + H_OFF);
    float*  stg0 = (float*)(smem + STG_OFF);            // 32x256 f32
    float*  stg1 = stg0 + 32 * NNODE;
    bf16_t* agg0 = (bf16_t*)(smem + AGG_OFF);           // 32x64 bf16
    bf16_t* agg1 = agg0 + 32 * DIM;
    float*  y0   = (float*)(smem + Y0_OFF);

    const int b    = blockIdx.x;
    const int tid  = threadIdx.x;
    const int wave = tid >> 5;
    const int lane = tid & 31;
    const int sp   = wave >> 2;      // strip-in-pair (0/1)
    const int nt   = wave & 3;       // n-tile
    const float* adjb = adj + (size_t)b * NNODE * NNODE;
    const float* vecb = vec + (size_t)b * NNODE * FDIM;

    // Kick the TDM for the first adj strip-pair (rows 0..31) before doing anything else.
    if (tid < 32) tdm_load_adj_tile(adjb, stg0, 32);

    // init: vec (256x63 f32) -> x0 (256x64 bf16, col 63 = 0), overlapped with the DMA
    {
        const float* vr = vecb + (size_t)tid * FDIM;
        bf16_t* xr = x0 + tid * DIM;
        for (int k = 0; k < FDIM; ++k) xr[k] = (bf16_t)vr[k];
        xr[FDIM] = (bf16_t)0.0f;
    }
    if (tid < 32) __builtin_amdgcn_s_wait_tensorcnt(0);
    __syncthreads();

    // ---- layer 1: 8 iterations x 2 strips; TDM double-buffers the adj stream ----
    for (int i = 0; i < 8; ++i) {
        float*  scur = (i & 1) ? stg1 : stg0;
        bf16_t* acur = (i & 1) ? agg1 : agg0;
        if ((tid < 32) && (i + 1 < 8)) {
            float* snxt = ((i + 1) & 1) ? stg1 : stg0;
            tdm_load_adj_tile(adjb + (size_t)(i + 1) * 32 * NNODE, snxt, 32);
        }
        const int strip = 2 * i + sp;
        const int row0  = strip * 16;

        // stage A: agg tile = adj_strip @ x0   (K = 256, 8 bf16 WMMA steps)
        v8f acc = {};
        #pragma unroll
        for (int kt = 0; kt < 8; ++kt) {
            v16bf a = load_a_stage_f32(scur, sp * 16, kt * 32, lane);
            acc = wmma_bf16(a, load_b_lds(x0, kt * 32, nt * 16, lane), acc);
        }
        store_c_bf16(acur, sp * 16, nt * 16, acc, lane);

        if (tid < 32) __builtin_amdgcn_s_wait_tensorcnt(0);  // next tile landed
        __syncthreads();                                     // agg visible + stage swap safe

        // stage B: h = relu(agg @ w1_rel + x0 @ w1_root + b1)   (K = 64)
        v8f hacc = {};
        #pragma unroll
        for (int kt = 0; kt < 2; ++kt) {
            hacc = wmma_bf16(load_a_lds(acur, sp * 16, kt * 32, lane),
                             load_b_global_f32(w1r, kt * 32, nt * 16, FDIM - 1, lane), hacc);
            hacc = wmma_bf16(load_a_lds(x0, row0, kt * 32, lane),
                             load_b_global_f32(w1t, kt * 32, nt * 16, FDIM - 1, lane), hacc);
        }
        const float bias = b1[nt * 16 + (lane & 15)];
        #pragma unroll
        for (int v = 0; v < 8; ++v) hacc[v] = fmaxf(hacc[v] + bias, 0.0f);
        store_c_bf16(h, row0, nt * 16, hacc, lane);
    }
    __syncthreads();

    // ---- layer 2: readout only needs node 0 -> rows 0..15 only ----
    bf16_t* agg2 = agg0;  // 16x64 shared region, waves 0..3 fill one 16-col tile each
    if (wave < 4) {
        v8f acc = {};
        #pragma unroll
        for (int kt = 0; kt < 8; ++kt) {
            v16bf a = load_a_global_f32(adjb, 0, kt * 32, NNODE, lane);  // rows 0..15
            acc = wmma_bf16(a, load_b_lds(h, kt * 32, nt * 16, lane), acc);
        }
        store_c_bf16(agg2, 0, nt * 16, acc, lane);
    }
    __syncthreads();
    if (wave < 4) {
        v8f yacc = {};
        #pragma unroll
        for (int kt = 0; kt < 2; ++kt) {
            yacc = wmma_bf16(load_a_lds(agg2, 0, kt * 32, lane),
                             load_b_global_f32(w2r, kt * 32, nt * 16, DIM - 1, lane), yacc);
            yacc = wmma_bf16(load_a_lds(h, 0, kt * 32, lane),
                             load_b_global_f32(w2t, kt * 32, nt * 16, DIM - 1, lane), yacc);
        }
        const float bias = b2[nt * 16 + (lane & 15)];
        const float v0 = fmaxf(yacc[0] + bias, 0.0f);  // VGPR0, lanes 0..15 = row M=0
        if (lane < 16) y0[nt * 16 + lane] = v0;
    }
    __syncthreads();

    // ---- readout: out[b] = (h1[0] + y[0]) @ wf + bf ----
    if (tid < 2) {
        float acc = bfv[tid];
        for (int k = 0; k < DIM; ++k)
            acc += ((float)h[k] + y0[k]) * wf[k * 2 + tid];
        out[(size_t)b * 2 + tid] = acc;
    }
}

extern "C" void kernel_launch(void* const* d_in, const int* in_sizes, int n_in,
                              void* d_out, int out_size, void* d_ws, size_t ws_size,
                              hipStream_t stream) {
    const float* adj = (const float*)d_in[0];
    const float* vec = (const float*)d_in[1];
    const float* w1r = (const float*)d_in[2];
    const float* w1t = (const float*)d_in[3];
    const float* b1  = (const float*)d_in[4];
    const float* w2r = (const float*)d_in[5];
    const float* w2t = (const float*)d_in[6];
    const float* b2  = (const float*)d_in[7];
    const float* wf  = (const float*)d_in[8];
    const float* bfv = (const float*)d_in[9];
    float* out = (float*)d_out;

    const int B = in_sizes[0] / (NNODE * NNODE);

    // 136 KB dynamic LDS (> default 64 KB cap); WGP has 320 KB -> 2 blocks/WGP.
    (void)hipFuncSetAttribute((const void*)gnn2_tdm_wmma_kernel,
                              hipFuncAttributeMaxDynamicSharedMemorySize, SMEM_BYTES);

    gnn2_tdm_wmma_kernel<<<dim3(B), dim3(THREADS), SMEM_BYTES, stream>>>(
        adj, vec, w1r, w1t, b1, w2r, w2t, b2, wf, bfv, out);
}
